// STICKYKVCache_LayerWise_80831284510823
// MI455X (gfx1250) — compile-verified, matrix operand order checked by reading, not run on hardware
//
#include <hip/hip_runtime.h>

typedef __attribute__((ext_vector_type(16))) _Float16 v16h;
typedef __attribute__((ext_vector_type(8)))  float    v8f;

#define HEADS 32
#define SEQ   4096
#define DHEAD 128
#define OMEGA 32
#define SINK  4
#define NWIN  127   // (SEQ - SINK) / OMEGA
#define QCNT  31    // NWIN / 4
#define QMAXF 255.0f

// -----------------------------------------------------------------------------
// Kernel A: colsum[h,k] = sum_q attn[h,q,k] via WMMA with an all-ones A matrix.
// Each wave owns a 16-column k-tile; 128 WMMA accumulations cover 4096 q-rows.
// With A==1, C[m,n] = sum_k B[k,n] independent of B's internal K permutation,
// so any disjoint assignment of 32 q-rows to (lane, half) is correct.
// -----------------------------------------------------------------------------
__global__ __launch_bounds__(128) void colsum_wmma_kernel(
    const float* __restrict__ attn, float* __restrict__ colsum) {
  const int wave = threadIdx.x >> 5;           // 0..3
  const int lane = threadIdx.x & 31;
  const int head = blockIdx.x;                 // 0..31
  const int tile = blockIdx.y * 4 + wave;      // 0..255
  const int k0   = tile * 16;
  const int col  = k0 + (lane & 15);
  const int rsel = (lane >> 4) * 16;           // lanes 16..31 take rows q+16..q+31

  v16h a;
  #pragma unroll
  for (int i = 0; i < 16; ++i) a[i] = (_Float16)1.0f;

  v8f c = {};

  const float* base = attn + ((size_t)head * SEQ) * SEQ + col;
  for (int q = 0; q < SEQ; q += 32) {
    const float* p = base + (size_t)(q + rsel) * SEQ;
    if (q + 32 < SEQ) __builtin_prefetch(p + (size_t)32 * SEQ, 0, 1);
    v16h b;
    #pragma unroll
    for (int j = 0; j < 16; ++j) {
      b[j] = (_Float16)p[(size_t)j * SEQ];
    }
    // 8 args: (neg_a, A, neg_b, B, c_mod, C, reuse_a, reuse_b)
    c = __builtin_amdgcn_wmma_f32_16x16x32_f16(false, a, false, b,
                                               (short)0, c, false, false);
  }
  // C layout: VGPR0 holds row M=0 (lanes 0-15, N=lane). All rows equal colsum.
  if (lane < 16) colsum[(size_t)head * SEQ + k0 + lane] = c[0];
}

// -----------------------------------------------------------------------------
// Kernel B: per-head window scores + stable bottom-QCNT selection.
// Matches jax.lax.top_k(-scores, QCNT): ascending score, ties -> lower index.
// -----------------------------------------------------------------------------
__global__ __launch_bounds__(128) void select_losers_kernel(
    const float* __restrict__ colsum, int* __restrict__ losers) {
  __shared__ float ws[NWIN];
  const int h = blockIdx.x;
  const int t = threadIdx.x;
  if (t < NWIN) {
    const float* p = colsum + (size_t)h * SEQ + SINK + t * OMEGA;
    float s = 0.f;
    #pragma unroll
    for (int i = 0; i < OMEGA; ++i) s += p[i];
    ws[t] = s;
  }
  __syncthreads();
  if (t < NWIN) {
    const float s = ws[t];
    int rank = 0;
    for (int j = 0; j < NWIN; ++j) {
      const float sj = ws[j];
      rank += (sj < s) || (sj == s && j < t);
    }
    if (rank < QCNT) losers[h * QCNT + rank] = t;
  }
}

// -----------------------------------------------------------------------------
// Kernel C: gather evicted windows, per-(head,window) asymmetric 8-bit
// quantize + dequantize for K then V. One block per (head, loser slot).
// -----------------------------------------------------------------------------
__global__ __launch_bounds__(256) void gather_quant_kernel(
    const float* __restrict__ past_k, const float* __restrict__ past_v,
    const int* __restrict__ losers, float* __restrict__ out) {
  __shared__ float smin[256];
  __shared__ float smax[256];
  const int h   = blockIdx.x;   // 0..31
  const int w   = blockIdx.y;   // 0..30
  const int tid = threadIdx.x;
  const int wid  = losers[h * QCNT + w];
  const int row0 = wid * OMEGA + SINK;

  const float* srcs[2] = { past_k, past_v };
  for (int s = 0; s < 2; ++s) {
    const float* src = srcs[s];
    float vals[16];
    float mn = 3.402823466e38f, mx = -3.402823466e38f;
    #pragma unroll
    for (int i = 0; i < 16; ++i) {
      const int e = tid + i * 256;        // 0..4095 = t*128 + d
      const int tt = e >> 7;
      const int d  = e & 127;
      int row = row0 + tt;
      if (row > SEQ - 1) row = SEQ - 1;   // jnp.clip (no-op here, kept for parity)
      const float x = src[((size_t)h * SEQ + row) * DHEAD + d];
      vals[i] = x;
      mn = fminf(mn, x);
      mx = fmaxf(mx, x);
    }
    smin[tid] = mn; smax[tid] = mx;
    __syncthreads();
    for (int off = 128; off > 0; off >>= 1) {
      if (tid < off) {
        smin[tid] = fminf(smin[tid], smin[tid + off]);
        smax[tid] = fmaxf(smax[tid], smax[tid + off]);
      }
      __syncthreads();
    }
    const float wmn = smin[0];
    const float wmx = smax[0];
    __syncthreads();   // safe reuse of smin/smax next pass
    const float scale = (wmx - wmn) / QMAXF + 1e-8f;
    const size_t obase =
        ((size_t)s * HEADS * QCNT + (size_t)h * QCNT + w) * (size_t)(OMEGA * DHEAD);
    #pragma unroll
    for (int i = 0; i < 16; ++i) {
      const int e = tid + i * 256;
      float q = rintf((vals[i] - wmn) / scale);   // round half-to-even == jnp.round
      q = fminf(fmaxf(q, 0.0f), QMAXF);
      out[obase + e] = q * scale + wmn;
    }
    __syncthreads();
  }
}

// -----------------------------------------------------------------------------
// Launch: d_in = {past_k, past_v, attn_score_cache}, all fp32.
// d_ws layout: [0, 512KB) colsum f32[32][4096]; then losers int[32][31].
// -----------------------------------------------------------------------------
extern "C" void kernel_launch(void* const* d_in, const int* in_sizes, int n_in,
                              void* d_out, int out_size, void* d_ws, size_t ws_size,
                              hipStream_t stream) {
  const float* past_k = (const float*)d_in[0];
  const float* past_v = (const float*)d_in[1];
  const float* attn   = (const float*)d_in[2];
  float* out    = (float*)d_out;
  float* colsum = (float*)d_ws;
  int*   losers = (int*)((char*)d_ws + (size_t)HEADS * SEQ * sizeof(float));

  dim3 gA(HEADS, SEQ / 64);     // 4 waves/block * 16 cols/wave = 64 cols/block
  colsum_wmma_kernel<<<gA, 128, 0, stream>>>(attn, colsum);

  select_losers_kernel<<<HEADS, 128, 0, stream>>>(colsum, losers);

  dim3 gC(HEADS, QCNT);
  gather_quant_kernel<<<gC, 256, 0, stream>>>(past_k, past_v, losers, out);
}
